// VisionMamba_21827023798550
// MI455X (gfx1250) — matrix operand order repaired and checked by reading, not compiled, not run on hardware
//
#include <hip/hip_runtime.h>
#include <hip/hip_bf16.h>
#include <cstdint>

#ifndef USE_ASYNC_LDS
#define USE_ASYNC_LDS 1
#endif

typedef __attribute__((ext_vector_type(16))) __bf16 v16bf;
typedef __attribute__((ext_vector_type(8)))  float  v8f;

namespace {
constexpr int BATCH = 32, DEPTH = 24, Dm = 384, Sm = 16, Rm = 24, Hm = 1024;
constexpr int IMG = 224, PP = 16, G14 = 14, Lm = 196;
constexpr int Mrows = BATCH * Lm;            // 6272 = 49 * 128
}

// ---------------------------------------------------------------------------
// bf16 WMMA GEMM:  C[M,N] = A[M,K] @ W[N,K]^T (+ bias[N])
// Block (256 thr = 8 waves) computes a 128x64 tile; the 64-col weight tile is
// staged through LDS per 64-wide K chunk (async global->LDS), shared by all 8
// waves. Each wave: 16x64 tile = 4 f32 accumulators, A frag reused across 4
// WMMAs, B frags from LDS (padded stride 144B -> conflict-free ds_load_b128).
// M % 128 == 0, N % 64 == 0, K % 32 == 0.
// ---------------------------------------------------------------------------
__global__ __launch_bounds__(256) void gemm_bf16_wmma(
    const __bf16* __restrict__ A, const __bf16* __restrict__ W,
    float* __restrict__ Cout, const float* __restrict__ bias,
    int Mr, int N, int Kd)
{
    constexpr int LSTR = 72;                         // bf16 elems; 144B row
    __shared__ __align__(16) __bf16 sB[64 * LSTR];   // 9216 B

    const int wave = threadIdx.x >> 5;
    const int lane = threadIdx.x & 31;
    const int tilesN = N >> 6;
    const int bm = (int)blockIdx.x / tilesN;
    const int bn = (int)blockIdx.x % tilesN;
    const int row0 = (bm << 7) + (wave << 4);
    const int n0   = bn << 6;

    const int half = lane >> 4;                      // 0 or 1
    const int l16  = lane & 15;

    const __bf16* Arow = A + (size_t)(row0 + l16) * Kd;

    v8f acc[4];
    acc[0] = v8f{}; acc[1] = v8f{}; acc[2] = v8f{}; acc[3] = v8f{};

    for (int kc = 0; kc < Kd; kc += 64) {
        const int kb = (Kd - kc < 64) ? (Kd - kc) : 64;   // 32 or 64
        const int segPerRow = kb >> 3;                    // 16B segments/row
        const int nseg = 64 * segPerRow;

        __syncthreads();                                  // tile consumed
        for (int s = threadIdx.x; s < nseg; s += 256) {
            const int r = s / segPerRow;
            const int c = s % segPerRow;
            const __bf16* gp = W + (size_t)(n0 + r) * Kd + kc + c * 8;
            __bf16*       lp = &sB[r * LSTR + c * 8];
#if USE_ASYNC_LDS
            unsigned long long ga = (unsigned long long)(uintptr_t)gp;
            unsigned           lo = (unsigned)(uintptr_t)lp;
            asm volatile("global_load_async_to_lds_b128 %0, %1, off"
                         :: "v"(lo), "v"(ga) : "memory");
#else
            *(uint4*)lp = *(const uint4*)gp;
#endif
        }
#if USE_ASYNC_LDS
        asm volatile("s_wait_asynccnt 0x0" ::: "memory");
#endif
        __syncthreads();                                  // tile ready

        __builtin_prefetch(Arow + kc + 64, 0, 3);         // next A chunk

        for (int ks = 0; ks < kb; ks += 32) {
            // A fragment (global): lane<16 -> row m, K +0..15;
            //                      lane>=16 -> K +16..31
            union { unsigned int u[8]; v16bf v; } a;
            const unsigned int* ap =
                (const unsigned int*)(Arow + kc + ks + half * 16);
            #pragma unroll
            for (int j = 0; j < 4; ++j) { a.u[j] = ap[j]; a.u[4 + j] = ap[4 + j]; }

            #pragma unroll
            for (int nt = 0; nt < 4; ++nt) {
                // B fragment (LDS): lane<16 -> col n, K +0..15 ; lane>=16 -> +16..31
                union { unsigned int u[8]; v16bf v; } b;
                const uint4* bp = (const uint4*)
                    &sB[(nt * 16 + l16) * LSTR + ks + half * 16];
                const uint4 b0 = bp[0], b1 = bp[1];
                b.u[0] = b0.x; b.u[1] = b0.y; b.u[2] = b0.z; b.u[3] = b0.w;
                b.u[4] = b1.x; b.u[5] = b1.y; b.u[6] = b1.z; b.u[7] = b1.w;

                acc[nt] = __builtin_amdgcn_wmma_f32_16x16x32_bf16(
                    false, a.v, false, b.v, (short)0, acc[nt], false, false);
            }
        }
    }

    #pragma unroll
    for (int nt = 0; nt < 4; ++nt) {
        const int col = n0 + nt * 16 + l16;
        const float bv = bias ? bias[col] : 0.f;
        #pragma unroll
        for (int r = 0; r < 8; ++r) {
            const int row = row0 + r + half * 8;   // C/D: vgpr r -> M=r (+8 hi half)
            Cout[(size_t)row * N + col] = acc[nt][r] + bv;
        }
    }
}

// ---------------------------------------------------------------------------
// LayerNorm over D=384, fp32 in -> bf16 out. One wave per row.
// ---------------------------------------------------------------------------
__global__ __launch_bounds__(256) void layernorm_bf16(
    const float* __restrict__ X, const float* __restrict__ w,
    const float* __restrict__ b, __bf16* __restrict__ Out, int Mr)
{
    const int wave = threadIdx.x >> 5;
    const int lane = threadIdx.x & 31;
    const int row  = blockIdx.x * (blockDim.x >> 5) + wave;
    if (row >= Mr) return;
    const float* xr = X + (size_t)row * Dm;
    float s = 0.f, s2 = 0.f;
    for (int i = lane; i < Dm; i += 32) { float v = xr[i]; s += v; s2 += v * v; }
    #pragma unroll
    for (int off = 16; off > 0; off >>= 1) {
        s  += __shfl_xor(s,  off, 32);
        s2 += __shfl_xor(s2, off, 32);
    }
    const float mu  = s / (float)Dm;
    const float var = s2 / (float)Dm - mu * mu;
    const float inv = rsqrtf(var + 1e-5f);
    __bf16* orow = Out + (size_t)row * Dm;
    for (int i = lane; i < Dm; i += 32)
        orow[i] = (__bf16)((xr[i] - mu) * inv * w[i] + b[i]);
}

// ---------------------------------------------------------------------------
// Depthwise causal conv1d (K=4, pad-left 3) + SiLU. Reads xi from XZ[:, :384].
// ---------------------------------------------------------------------------
__global__ void conv_silu_kernel(
    const float* __restrict__ XZ, const float* __restrict__ cw,
    const float* __restrict__ cb, float* __restrict__ XIo,
    __bf16* __restrict__ XIbf)
{
    const int idx = blockIdx.x * blockDim.x + threadIdx.x;
    if (idx >= Mrows * Dm) return;
    const int d   = idx % Dm;
    const int row = idx / Dm;
    const int l   = row % Lm;
    float acc = cb[d];
    #pragma unroll
    for (int j = 0; j < 4; ++j) {
        const int ls = l - 3 + j;
        if (ls >= 0) acc += XZ[(size_t)(row - 3 + j) * (2 * Dm) + d] * cw[d * 4 + j];
    }
    const float s = acc / (1.f + __expf(-acc));
    XIo[idx]  = s;
    XIbf[idx] = (__bf16)s;
}

// x_dbl[:, :24] -> bf16 [M,32] (zero-padded K for dt GEMM)
__global__ void build_dtin_kernel(const float* __restrict__ XDBL,
                                  __bf16* __restrict__ DTIN)
{
    const int idx = blockIdx.x * blockDim.x + threadIdx.x;
    if (idx >= Mrows * 32) return;
    const int c = idx & 31;
    const int row = idx >> 5;
    const float v = (c < Rm) ? XDBL[(size_t)row * 64 + c] : 0.f;
    DTIN[idx] = (__bf16)v;
}

// ---------------------------------------------------------------------------
// Selective scan: sequential over L, parallel over (batch, d-chunk).
// grid = (BATCH, 3), block = 128 -> 384 d's per batch. S=16 state in regs.
// Fuses softplus(dt)+dt_b, x_dbl B/C broadcast via LDS, D-skip, SiLU(z) gate,
// writes gated y as bf16 (input to out_proj GEMM).
// ---------------------------------------------------------------------------
__global__ __launch_bounds__(128) void scan_kernel(
    const float* __restrict__ DTRAW, const float* __restrict__ dt_b,
    const float* __restrict__ XI,    const float* __restrict__ XDBL,
    const float* __restrict__ A_log, const float* __restrict__ D_p,
    const float* __restrict__ XZ,    __bf16* __restrict__ YBF)
{
    const int b = blockIdx.x;
    const int d = blockIdx.y * 128 + threadIdx.x;
    __shared__ float Bs[Sm], Cs[Sm];

    float Ac[Sm], hs[Sm];
    #pragma unroll
    for (int s = 0; s < Sm; ++s) {
        Ac[s] = -__expf(A_log[(size_t)d * Sm + s]);
        hs[s] = 0.f;
    }
    const float dtb = dt_b[d];
    const float Dd  = D_p[d];

    for (int l = 0; l < Lm; ++l) {
        const size_t row = (size_t)b * Lm + l;
        __syncthreads();
        if (threadIdx.x < 32) {
            const int t = threadIdx.x;
            if (t < 16) Bs[t]      = XDBL[row * 64 + Rm + t];
            else        Cs[t - 16] = XDBL[row * 64 + Rm + Sm + (t - 16)];
        }
        __syncthreads();
        const float dtl = DTRAW[row * Dm + d] + dtb;
        const float dt  = (dtl > 20.f) ? dtl : log1pf(__expf(dtl));
        const float xiv = XI[row * Dm + d];
        const float dtxi = dt * xiv;
        float y = 0.f;
        #pragma unroll
        for (int s = 0; s < Sm; ++s) {
            const float dA = __expf(dt * Ac[s]);
            hs[s] = dA * hs[s] + dtxi * Bs[s];
            y += hs[s] * Cs[s];
        }
        y += Dd * xiv;
        const float zv = XZ[row * (2 * Dm) + Dm + d];
        y *= zv / (1.f + __expf(-zv));        // y * silu(z)
        YBF[row * Dm + d] = (__bf16)y;
    }
}

// SwiGLU combine: hid = silu(u1 + b1) * (u2 + b2) -> bf16
__global__ void swiglu_kernel(const float* __restrict__ U1,
                              const float* __restrict__ U2,
                              const float* __restrict__ b1,
                              const float* __restrict__ b2,
                              __bf16* __restrict__ HID)
{
    const int idx = blockIdx.x * blockDim.x + threadIdx.x;
    if (idx >= Mrows * Hm) return;
    const int c = idx % Hm;
    const float u1 = U1[idx] + b1[c];
    const float u2 = U2[idx] + b2[c];
    HID[idx] = (__bf16)((u1 / (1.f + __expf(-u1))) * u2);
}

__global__ void add_inplace_kernel(float* __restrict__ dst,
                                   const float* __restrict__ src, int n)
{
    const int idx = blockIdx.x * blockDim.x + threadIdx.x;
    if (idx < n) dst[idx] += src[idx];
}

__global__ void copy_f32_kernel(float* __restrict__ dst,
                                const float* __restrict__ src, int n)
{
    const int idx = blockIdx.x * blockDim.x + threadIdx.x;
    if (idx < n) dst[idx] = src[idx];
}

// f32 [srcR, srcC] -> bf16 [dstR, dstC], zero padded
__global__ void convert_pad_kernel(const float* __restrict__ src,
                                   __bf16* __restrict__ dst,
                                   int srcR, int srcC, int dstR, int dstC)
{
    const int idx = blockIdx.x * blockDim.x + threadIdx.x;
    if (idx >= dstR * dstC) return;
    const int c = idx % dstC, r = idx / dstC;
    const float v = (r < srcR && c < srcC) ? src[(size_t)r * srcC + c] : 0.f;
    dst[idx] = (__bf16)v;
}

// im2col of 16x16 patches -> bf16 [M, 768]
__global__ void im2col_kernel(const float* __restrict__ x,
                              __bf16* __restrict__ out)
{
    const int idx = blockIdx.x * blockDim.x + threadIdx.x;
    if (idx >= Mrows * 768) return;
    const int col = idx % 768, row = idx / 768;
    const int b = row / Lm, l = row % Lm;
    const int c = col >> 8, rem = col & 255, iy = rem >> 4, ix = rem & 15;
    const int yy = (l / G14) * PP + iy, xx = (l % G14) * PP + ix;
    out[idx] = (__bf16)x[(((size_t)b * 3 + c) * IMG + yy) * IMG + xx];
}

// ---------------------------------------------------------------------------
extern "C" void kernel_launch(void* const* d_in, const int* in_sizes, int n_in,
                              void* d_out, int out_size, void* d_ws, size_t ws_size,
                              hipStream_t stream)
{
    const float* x       = (const float*)d_in[0];
    const float* patch_w = (const float*)d_in[1];
    const float* patch_b = (const float*)d_in[2];
    const float* ln1_w   = (const float*)d_in[3];
    const float* ln1_b   = (const float*)d_in[4];
    const float* in_w    = (const float*)d_in[5];
    const float* conv_w  = (const float*)d_in[6];
    const float* conv_b  = (const float*)d_in[7];
    const float* xproj_w = (const float*)d_in[8];
    const float* dt_w    = (const float*)d_in[9];
    const float* dt_b    = (const float*)d_in[10];
    const float* A_log   = (const float*)d_in[11];
    const float* D_p     = (const float*)d_in[12];
    const float* out_w   = (const float*)d_in[13];
    const float* ln2_w   = (const float*)d_in[14];
    const float* ln2_b   = (const float*)d_in[15];
    const float* w1_w    = (const float*)d_in[16];
    const float* w1_b    = (const float*)d_in[17];
    const float* w2_w    = (const float*)d_in[18];
    const float* w2_b    = (const float*)d_in[19];
    const float* w3_w    = (const float*)d_in[20];
    const float* w3_b    = (const float*)d_in[21];

    // ---- workspace layout ----
    char* ws = (char*)d_ws;
    size_t p = 0;
    auto carve = [&](size_t bytes) { size_t o = p; p += (bytes + 255) & ~(size_t)255; return o; };
    float*  H_    = (float*) (ws + carve((size_t)Mrows * Dm * 4));  // residual stream
    __bf16* ABF   = (__bf16*)(ws + carve((size_t)Mrows * Hm * 2));  // GEMM A (also patches/hid)
    float*  B1    = (float*) (ws + carve((size_t)Mrows * Hm * 4));  // xz / u1 / mlp out
    float*  B2    = (float*) (ws + carve((size_t)Mrows * Hm * 4));  // u2 / out_proj out
    float*  XI    = (float*) (ws + carve((size_t)Mrows * Dm * 4));  // conv+silu out f32
    __bf16* XIBF  = (__bf16*)(ws + carve((size_t)Mrows * Dm * 2));  // xi bf16 / gated y bf16
    float*  XDBL  = (float*) (ws + carve((size_t)Mrows * 64 * 4));  // x_dbl (padded N=64)
    __bf16* DTIN  = (__bf16*)(ws + carve((size_t)Mrows * 32 * 2));  // dt GEMM A (padded K=32)
    float*  DTRAW = (float*) (ws + carve((size_t)Mrows * Dm * 4));  // dt pre-softplus
    __bf16* WBF   = (__bf16*)(ws + carve((size_t)1658880 * 2));     // per-layer bf16 weights
    (void)ws_size; (void)n_in; (void)in_sizes; (void)out_size;

    // per-layer bf16 weight slots (element offsets)
    __bf16* Win  = WBF + 0;        // 768 x 384
    __bf16* Wxp  = WBF + 294912;   //  64 x 384 (rows 56..63 zero)
    __bf16* Wdt  = WBF + 319488;   // 384 x  32 (cols 24..31 zero)
    __bf16* Wout = WBF + 331776;   // 384 x 384
    __bf16* W1   = WBF + 479232;   // 1024 x 384
    __bf16* W2   = WBF + 872448;   // 1024 x 384
    __bf16* W3   = WBF + 1265664;  // 384 x 1024

    auto blocksOf = [](size_t n, int bs) { return (unsigned)((n + bs - 1) / bs); };
    auto gemm = [&](const __bf16* Ap, const __bf16* Wp, float* Cp,
                    const float* biasp, int Mr, int N, int Kd) {
        const int blocks = (Mr / 128) * (N / 64);
        gemm_bf16_wmma<<<dim3(blocks), dim3(256), 0, stream>>>(
            Ap, Wp, Cp, biasp, Mr, N, Kd);
    };
    auto cvt = [&](const float* src, __bf16* dst, int sr, int sc, int dr, int dc) {
        convert_pad_kernel<<<blocksOf((size_t)dr * dc, 256), 256, 0, stream>>>(
            src, dst, sr, sc, dr, dc);
    };

    // ---- patch embedding: h = im2col(x) @ patch_w^T + patch_b ----
    im2col_kernel<<<blocksOf((size_t)Mrows * 768, 256), 256, 0, stream>>>(x, ABF);
    cvt(patch_w, Win, Dm, 768, Dm, 768);
    gemm(ABF, Win, H_, patch_b, Mrows, Dm, 768);

    // ---- 24 layers ----
    for (int l = 0; l < DEPTH; ++l) {
        const float* ln1w = ln1_w + (size_t)l * Dm;
        const float* ln1b = ln1_b + (size_t)l * Dm;
        const float* inw  = in_w  + (size_t)l * 2 * Dm * Dm;
        const float* cw   = conv_w+ (size_t)l * Dm * 4;
        const float* cb   = conv_b+ (size_t)l * Dm;
        const float* xpw  = xproj_w + (size_t)l * (Rm + 2 * Sm) * Dm;
        const float* dtw  = dt_w  + (size_t)l * Dm * Rm;
        const float* dtb  = dt_b  + (size_t)l * Dm;
        const float* alog = A_log + (size_t)l * Dm * Sm;
        const float* dp   = D_p   + (size_t)l * Dm;
        const float* ow   = out_w + (size_t)l * Dm * Dm;
        const float* ln2w = ln2_w + (size_t)l * Dm;
        const float* ln2b = ln2_b + (size_t)l * Dm;
        const float* a1w  = w1_w + (size_t)l * Hm * Dm;
        const float* a1b  = w1_b + (size_t)l * Hm;
        const float* a2w  = w2_w + (size_t)l * Hm * Dm;
        const float* a2b  = w2_b + (size_t)l * Hm;
        const float* a3w  = w3_w + (size_t)l * Dm * Hm;
        const float* a3b  = w3_b + (size_t)l * Dm;

        // weight conversion (per layer, into reused staging)
        cvt(inw, Win, 2 * Dm, Dm, 2 * Dm, Dm);
        cvt(xpw, Wxp, Rm + 2 * Sm, Dm, 64, Dm);
        cvt(dtw, Wdt, Dm, Rm, Dm, 32);
        cvt(ow,  Wout, Dm, Dm, Dm, Dm);
        cvt(a1w, W1, Hm, Dm, Hm, Dm);
        cvt(a2w, W2, Hm, Dm, Hm, Dm);
        cvt(a3w, W3, Dm, Hm, Dm, Hm);

        // --- mamba path ---
        layernorm_bf16<<<Mrows / 8, 256, 0, stream>>>(H_, ln1w, ln1b, ABF, Mrows);
        gemm(ABF, Win, B1, nullptr, Mrows, 2 * Dm, Dm);               // xz
        conv_silu_kernel<<<blocksOf((size_t)Mrows * Dm, 256), 256, 0, stream>>>(
            B1, cw, cb, XI, XIBF);                                    // xi
        gemm(XIBF, Wxp, XDBL, nullptr, Mrows, 64, Dm);                // x_dbl
        build_dtin_kernel<<<blocksOf((size_t)Mrows * 32, 256), 256, 0, stream>>>(XDBL, DTIN);
        gemm(DTIN, Wdt, DTRAW, nullptr, Mrows, Dm, 32);               // dt pre-softplus
        scan_kernel<<<dim3(BATCH, Dm / 128), dim3(128), 0, stream>>>(
            DTRAW, dtb, XI, XDBL, alog, dp, B1, XIBF);                // gated y (bf16)
        gemm(XIBF, Wout, B2, nullptr, Mrows, Dm, Dm);                 // out_proj
        add_inplace_kernel<<<blocksOf((size_t)Mrows * Dm, 256), 256, 0, stream>>>(
            H_, B2, Mrows * Dm);

        // --- MLP path ---
        layernorm_bf16<<<Mrows / 8, 256, 0, stream>>>(H_, ln2w, ln2b, ABF, Mrows);
        gemm(ABF, W1, B1, nullptr, Mrows, Hm, Dm);                    // u1
        gemm(ABF, W2, B2, nullptr, Mrows, Hm, Dm);                    // u2
        swiglu_kernel<<<blocksOf((size_t)Mrows * Hm, 256), 256, 0, stream>>>(
            B1, B2, a1b, a2b, ABF);                                   // hid (bf16)
        gemm(ABF, W3, B1, a3b, Mrows, Dm, Hm);                        // mlp out + b3
        add_inplace_kernel<<<blocksOf((size_t)Mrows * Dm, 256), 256, 0, stream>>>(
            H_, B1, Mrows * Dm);
    }

    // ---- output ----
    copy_f32_kernel<<<blocksOf((size_t)Mrows * Dm, 256), 256, 0, stream>>>(
        (float*)d_out, H_, Mrows * Dm);
}
